// StateGNNEncoderConv_39702677684856
// MI455X (gfx1250) — matrix operand order repaired and checked by reading, not compiled, not run on hardware
//
#include <hip/hip_runtime.h>

typedef __attribute__((ext_vector_type(16))) _Float16 v16h;
typedef __attribute__((ext_vector_type(8)))  _Float16 v8h;
typedef __attribute__((ext_vector_type(4)))  _Float16 v4h;
typedef __attribute__((ext_vector_type(8)))  float    v8f;

#define NG   50000
#define NS   50000
#define NE   800000
#define HC   256
#define OUTC 64
#define KHOP 10   // K+1 = 11 weight matrices per TAGConv

#define BSTRIDE 48   // padded LDS row stride (halves) for the B slab

// ---------------------------------------------------------------------------
// Elementwise / staging kernels
// ---------------------------------------------------------------------------
__global__ void zero_f32(float* __restrict__ p, long long n) {
  long long i = (long long)blockIdx.x * blockDim.x + threadIdx.x;
  if (i < n) p[i] = 0.0f;
}

__global__ void relu_f32(float* __restrict__ p, long long n) {
  long long i = (long long)blockIdx.x * blockDim.x + threadIdx.x;
  if (i < n) p[i] = fmaxf(p[i], 0.0f);
}

__global__ void init_bias(float* __restrict__ y, const float* __restrict__ b,
                          long long mn, int n) {
  long long i = (long long)blockIdx.x * blockDim.x + threadIdx.x;
  if (i < mn) y[i] = b[i % n];
}

// vectorized f32 -> f16 (n must be a multiple of 4; all staged buffers are M*256)
__global__ void cvt_f16(const float* __restrict__ s, _Float16* __restrict__ d,
                        long long n4) {
  long long i = (long long)blockIdx.x * blockDim.x + threadIdx.x;
  if (i >= n4) return;
  float4 v = *(const float4*)(s + i * 4);
  v4h h = { (_Float16)v.x, (_Float16)v.y, (_Float16)v.z, (_Float16)v.w };
  *(v4h*)(d + i * 4) = h;
}

// W [K,N] f32 row-major  ->  WT [N,K] f16 (column-of-W contiguous)
__global__ void cvtT_f16(const float* __restrict__ w, _Float16* __restrict__ wt,
                         int k, int n) {
  long long i = (long long)blockIdx.x * blockDim.x + threadIdx.x;
  if (i < (long long)k * n) {
    int kk = (int)(i / n), nn = (int)(i % n);
    wt[(long long)nn * k + kk] = (_Float16)w[i];
  }
}

// ---------------------------------------------------------------------------
// Graph kernels
// ---------------------------------------------------------------------------
__global__ void degree_k(const int* __restrict__ edge, float* __restrict__ deg,
                         int nE) {
  int e = blockIdx.x * blockDim.x + threadIdx.x;
  if (e < nE) atomicAdd(&deg[edge[nE + e]], 1.0f);
}

__global__ void rsqrt_k(float* __restrict__ d, int n) {
  int i = blockIdx.x * blockDim.x + threadIdx.x;
  if (i < n) { float v = d[i]; d[i] = v > 0.0f ? rsqrtf(v) : 0.0f; }
}

// one hop of normalized propagation, C = 256 (float4 per thread)
__global__ void prop256(const float* __restrict__ h, const int* __restrict__ edge,
                        const float* __restrict__ dinv, float* __restrict__ ho,
                        int nE) {
  long long i = (long long)blockIdx.x * blockDim.x + threadIdx.x;
  if (i >= (long long)nE * (HC / 4)) return;
  int e = (int)(i >> 6);
  int c = ((int)i & 63) << 2;
  int r = edge[e], cd = edge[nE + e];
  float w = dinv[r] * dinv[cd];
  float4 v = *(const float4*)(h + (long long)r * HC + c);
  float* o = ho + (long long)cd * HC + c;
  atomicAdd(o + 0, w * v.x); atomicAdd(o + 1, w * v.y);
  atomicAdd(o + 2, w * v.z); atomicAdd(o + 3, w * v.w);
}

// one hop of normalized propagation, small C (5 or 6)
__global__ void prop_small(const float* __restrict__ h, const int* __restrict__ edge,
                           const float* __restrict__ dinv, float* __restrict__ ho,
                           int nE, int c) {
  long long i = (long long)blockIdx.x * blockDim.x + threadIdx.x;
  if (i >= (long long)nE * c) return;
  int e = (int)(i / c), cc = (int)(i % c);
  int r = edge[e], cd = edge[nE + e];
  atomicAdd(&ho[(long long)cd * c + cc], dinv[r] * dinv[cd] * h[(long long)r * c + cc]);
}

// SAGE neighbor sum (C = 256)
__global__ void sage_sum(const float* __restrict__ x, const int* __restrict__ edge,
                         float* __restrict__ agg, int nE) {
  long long i = (long long)blockIdx.x * blockDim.x + threadIdx.x;
  if (i >= (long long)nE * (HC / 4)) return;
  int e = (int)(i >> 6);
  int c = ((int)i & 63) << 2;
  int r = edge[e], cd = edge[nE + e];
  float4 v = *(const float4*)(x + (long long)r * HC + c);
  float* o = agg + (long long)cd * HC + c;
  atomicAdd(o + 0, v.x); atomicAdd(o + 1, v.y);
  atomicAdd(o + 2, v.z); atomicAdd(o + 3, v.w);
}

__global__ void mean_div(float* __restrict__ agg, const float* __restrict__ cnt,
                         long long mn) {
  long long i = (long long)blockIdx.x * blockDim.x + threadIdx.x;
  if (i < mn) agg[i] /= fmaxf(cnt[i >> 8], 1.0f);   // H == 256
}

// ---------------------------------------------------------------------------
// GEMMs
// ---------------------------------------------------------------------------
// Tiny-K GEMM (K = 5/6): Y[M,N] += X[M,K] @ W[K,N]
__global__ void gemm_small_acc(const float* __restrict__ x, const float* __restrict__ w,
                               float* __restrict__ y, int m, int k, int n) {
  long long i = (long long)blockIdx.x * blockDim.x + threadIdx.x;
  if (i >= (long long)m * n) return;
  int mm = (int)(i / n), nn = (int)(i % n);
  float s = 0.0f;
  for (int kk = 0; kk < k; ++kk) s += x[(long long)mm * k + kk] * w[kk * n + nn];
  y[i] += s;
}

// WMMA GEMM: Y[M,N] += X[M,K]_f16 @ WT[N,K]_f16^T   (f32 accumulate), K fixed
// at compile time (always 256 here) so the k-loop fully unrolls and all slab /
// fragment addresses become immediate offsets.
// 8 waves / block; each wave owns one 16-row tile x FOUR 16-col tiles.
// The 64-col x 32-K B slab is staged in LDS once per block per k-step and
// shared by all 8 waves (ds_load_b128); A fragment is reused across 4 WMMAs.
template <int K>
__global__ void __launch_bounds__(256)
wmma_gemm_acc(const _Float16* __restrict__ X, const _Float16* __restrict__ WT,
              float* __restrict__ Y, int M, int N) {
  __shared__ _Float16 bs[64 * BSTRIDE];     // 6 KB (padded rows)

  const int tid  = threadIdx.x;
  const int wave = tid >> 5;
  const int lane = tid & 31;
  const int m0 = (blockIdx.x * 8 + wave) * 16;
  const int n0 = blockIdx.y * 64;
  const int hi = lane >> 4;
  const int lr = lane & 15;

  // Clamp rows so every wave participates in the barriers (ragged tail);
  // stores are masked below. EXEC stays all-ones around every WMMA.
  int ma = m0 + lr; if (ma >= M) ma = M - 1;
  const _Float16* xrow = X + (long long)ma * K;

  // cooperative B stage: thread t loads 8 halves of column (n0+bn), k-off bk
  const int bn = tid >> 2;                  // 0..63
  const int bk = (tid & 3) * 8;             // 0,8,16,24
  const _Float16* wsrc = WT + (long long)(n0 + bn) * K + bk;
  _Float16* bdst = bs + bn * BSTRIDE + bk;

  v8f acc[4] = {};
#pragma unroll
  for (int k0 = 0; k0 < K; k0 += 32) {
    *(v8h*)(bdst) = *(const v8h*)(wsrc + k0);
    __syncthreads();

    // A 16x32 f16: lane (hi,lr) holds row lr, K = {k0+hi*8..+7, k0+16+hi*8..+7}
    v8h alo = *(const v8h*)(xrow + k0 + hi * 8);
    v8h ahi = *(const v8h*)(xrow + k0 + 16 + hi * 8);
    v16h a;
#pragma unroll
    for (int e = 0; e < 8; ++e) { a[e] = alo[e]; a[e + 8] = ahi[e]; }

    // fetch all four B fragments, then issue the four WMMAs
    v16h b[4];
#pragma unroll
    for (int nt = 0; nt < 4; ++nt) {
      const _Float16* bp = bs + (nt * 16 + lr) * BSTRIDE + hi * 8;
      v8h blo = *(const v8h*)(bp);
      v8h bhi = *(const v8h*)(bp + 16);
#pragma unroll
      for (int e = 0; e < 8; ++e) { b[nt][e] = blo[e]; b[nt][e + 8] = bhi[e]; }
    }
#pragma unroll
    for (int nt = 0; nt < 4; ++nt)
      acc[nt] = __builtin_amdgcn_wmma_f32_16x16x32_f16(false, a, false, b[nt],
                                                       (short)0, acc[nt], false, false);
    __syncthreads();
  }

#pragma unroll
  for (int nt = 0; nt < 4; ++nt) {
#pragma unroll
    for (int v = 0; v < 8; ++v) {
      int m = m0 + hi * 8 + v;              // C/D layout: VGPR v -> row hi*8+v
      if (m < M) Y[(long long)m * N + n0 + nt * 16 + lr] += acc[nt][v];
    }
  }
}

// ---------------------------------------------------------------------------
// Host-side orchestration helpers (all launches on `stream`)
// ---------------------------------------------------------------------------
static inline int cdiv(long long a, int b) { return (int)((a + b - 1) / b); }

static void run_wmma(const _Float16* x16, const _Float16* wt, float* y,
                     int m, int n, hipStream_t s) {
  dim3 g(cdiv(cdiv(m, 16), 8), n / 64);
  wmma_gemm_acc<HC><<<g, 256, 0, s>>>(x16, wt, y, m, n);   // K == 256 always
}

static void run_cvt(const float* src, _Float16* dst, long long n, hipStream_t s) {
  cvt_f16<<<cdiv(n / 4, 256), 256, 0, s>>>(src, dst, n / 4);
}

// TAGConv with small input width (5/6): out = relu(sum_k A^k x @ W[k] + b)
static void tag_small(const float* x, int cin, const int* edge, const float* dinv,
                      const float* W, const float* b, int m, float* out,
                      float* h0, float* h1, hipStream_t s) {
  long long mn = (long long)m * HC;
  init_bias<<<cdiv(mn, 256), 256, 0, s>>>(out, b, mn, HC);
  gemm_small_acc<<<cdiv(mn, 256), 256, 0, s>>>(x, W, out, m, cin, HC);
  const float* cur = x;
  for (int k = 1; k <= KHOP; ++k) {
    float* nxt = (k & 1) ? h0 : h1;
    long long hn = (long long)m * cin;
    zero_f32<<<cdiv(hn, 256), 256, 0, s>>>(nxt, hn);
    prop_small<<<cdiv((long long)NE * cin, 256), 256, 0, s>>>(cur, edge, dinv, nxt, NE, cin);
    gemm_small_acc<<<cdiv(mn, 256), 256, 0, s>>>(nxt, W + (long long)k * cin * HC,
                                                 out, m, cin, HC);
    cur = nxt;
  }
  relu_f32<<<cdiv(mn, 256), 256, 0, s>>>(out, mn);
}

// TAGConv H->H via WMMA
static void tag_big(const float* x, const int* edge, const float* dinv,
                    const float* W, const float* b, int m, float* out,
                    float* h0, float* h1, _Float16* x16, _Float16* w16,
                    hipStream_t s) {
  long long mn = (long long)m * HC;
  for (int k = 0; k <= KHOP; ++k)
    cvtT_f16<<<cdiv(HC * HC, 256), 256, 0, s>>>(W + (long long)k * HC * HC,
                                                w16 + (long long)k * HC * HC, HC, HC);
  init_bias<<<cdiv(mn, 256), 256, 0, s>>>(out, b, mn, HC);
  run_cvt(x, x16, mn, s);
  run_wmma(x16, w16, out, m, HC, s);
  const float* cur = x;
  for (int k = 1; k <= KHOP; ++k) {
    float* nxt = (k & 1) ? h0 : h1;
    zero_f32<<<cdiv(mn, 256), 256, 0, s>>>(nxt, mn);
    prop256<<<cdiv((long long)NE * (HC / 4), 256), 256, 0, s>>>(cur, edge, dinv, nxt, NE);
    run_cvt(nxt, x16, mn, s);
    run_wmma(x16, w16 + (long long)k * HC * HC, out, m, HC, s);
    cur = nxt;
  }
  relu_f32<<<cdiv(mn, 256), 256, 0, s>>>(out, mn);
}

// SAGEConv: out = relu(mean_agg(src) @ Wl + bl + dst @ Wr)
static void sage(const float* src, const float* dst, const int* edge,
                 const float* Wl, const float* bl, const float* Wr, float* out,
                 float* agg, float* cnt, _Float16* x16, _Float16* w16,
                 hipStream_t s) {
  long long mn = (long long)NS * HC;
  zero_f32<<<cdiv(mn, 256), 256, 0, s>>>(agg, mn);
  zero_f32<<<cdiv((long long)NS, 256), 256, 0, s>>>(cnt, NS);
  degree_k<<<cdiv(NE, 256), 256, 0, s>>>(edge, cnt, NE);
  sage_sum<<<cdiv((long long)NE * (HC / 4), 256), 256, 0, s>>>(src, edge, agg, NE);
  mean_div<<<cdiv(mn, 256), 256, 0, s>>>(agg, cnt, mn);
  init_bias<<<cdiv(mn, 256), 256, 0, s>>>(out, bl, mn, HC);
  cvtT_f16<<<cdiv(HC * HC, 256), 256, 0, s>>>(Wl, w16, HC, HC);
  run_cvt(agg, x16, mn, s);
  run_wmma(x16, w16, out, NS, HC, s);
  cvtT_f16<<<cdiv(HC * HC, 256), 256, 0, s>>>(Wr, w16, HC, HC);
  run_cvt(dst, x16, mn, s);
  run_wmma(x16, w16, out, NS, HC, s);
  relu_f32<<<cdiv(mn, 256), 256, 0, s>>>(out, mn);
}

// ---------------------------------------------------------------------------
extern "C" void kernel_launch(void* const* d_in, const int* in_sizes, int n_in,
                              void* d_out, int out_size, void* d_ws, size_t ws_size,
                              hipStream_t stream) {
  (void)in_sizes; (void)n_in; (void)out_size; (void)ws_size;
  const float* x_game  = (const float*)d_in[0];
  const float* x_state = (const float*)d_in[1];
  const int* e_gg   = (const int*)d_in[2];
  const int* e_ss   = (const int*)d_in[3];
  const int* e_hist = (const int*)d_in[4];
  const int* e_in   = (const int*)d_in[5];
  const float* W1  = (const float*)d_in[6];   const float* b1  = (const float*)d_in[7];
  const float* W12 = (const float*)d_in[8];   const float* b12 = (const float*)d_in[9];
  const float* W2  = (const float*)d_in[10];  const float* b2  = (const float*)d_in[11];
  const float* W22 = (const float*)d_in[12];  const float* b22 = (const float*)d_in[13];
  const float* Wl3  = (const float*)d_in[14]; const float* bl3  = (const float*)d_in[15];
  const float* Wr3  = (const float*)d_in[16];
  const float* Wl32 = (const float*)d_in[17]; const float* bl32 = (const float*)d_in[18];
  const float* Wr32 = (const float*)d_in[19];
  const float* Wl4  = (const float*)d_in[20]; const float* bl4  = (const float*)d_in[21];
  const float* Wr4  = (const float*)d_in[22];
  const float* Wl42 = (const float*)d_in[23]; const float* bl42 = (const float*)d_in[24];
  const float* Wr42 = (const float*)d_in[25];
  const float* Wlin = (const float*)d_in[26]; const float* blin = (const float*)d_in[27];
  float* out = (float*)d_out;

  // workspace carve-up
  char* ws = (char*)d_ws; size_t off = 0;
  auto carve = [&](size_t bytes) -> void* {
    void* p = ws + off; off += (bytes + 255) & ~(size_t)255; return p;
  };
  float*    gx     = (float*)carve((size_t)NG * HC * 4);
  float*    sx     = (float*)carve((size_t)NS * HC * 4);
  float*    acc    = (float*)carve((size_t)NG * HC * 4);
  float*    h0     = (float*)carve((size_t)NG * HC * 4);
  float*    h1     = (float*)carve((size_t)NG * HC * 4);
  _Float16* x16    = (_Float16*)carve((size_t)NG * HC * 2);
  _Float16* w16    = (_Float16*)carve((size_t)(KHOP + 1) * HC * HC * 2);
  float*    dinv_g = (float*)carve((size_t)NG * 4);
  float*    dinv_s = (float*)carve((size_t)NS * 4);
  float*    cnt    = (float*)carve((size_t)NS * 4);

  // gcn_norm degrees -> d^-1/2
  zero_f32<<<cdiv(NG, 256), 256, 0, stream>>>(dinv_g, NG);
  degree_k<<<cdiv(NE, 256), 256, 0, stream>>>(e_gg, dinv_g, NE);
  rsqrt_k<<<cdiv(NG, 256), 256, 0, stream>>>(dinv_g, NG);
  zero_f32<<<cdiv(NS, 256), 256, 0, stream>>>(dinv_s, NS);
  degree_k<<<cdiv(NE, 256), 256, 0, stream>>>(e_ss, dinv_s, NE);
  rsqrt_k<<<cdiv(NS, 256), 256, 0, stream>>>(dinv_s, NS);

  // game tower:  gx = relu(TAG(relu(TAG(x_game))))
  tag_small(x_game, 5, e_gg, dinv_g, W1, b1, NG, acc, h0, h1, stream);
  tag_big(acc, e_gg, dinv_g, W12, b12, NG, gx, h0, h1, x16, w16, stream);
  // state tower: sx = relu(TAG(relu(TAG(x_state))))
  tag_small(x_state, 6, e_ss, dinv_s, W2, b2, NS, acc, h0, h1, stream);
  tag_big(acc, e_ss, dinv_s, W22, b22, NS, sx, h0, h1, x16, w16, stream);

  // 4 SAGE layers (ping-pong sx <-> acc); agg reuses h0, second W reuses w16
  sage(gx, sx,  e_hist, Wl3,  bl3,  Wr3,  acc, h0, cnt, x16, w16, stream);
  sage(gx, acc, e_hist, Wl32, bl32, Wr32, sx,  h0, cnt, x16, w16, stream);
  sage(gx, sx,  e_in,   Wl4,  bl4,  Wr4,  acc, h0, cnt, x16, w16, stream);
  sage(gx, acc, e_in,   Wl42, bl42, Wr42, sx,  h0, cnt, x16, w16, stream);

  // final linear: out = sx @ Wlin + blin   [NS, OUTC]
  long long mo = (long long)NS * OUTC;
  cvtT_f16<<<cdiv(HC * OUTC, 256), 256, 0, stream>>>(Wlin, w16, HC, OUTC);
  init_bias<<<cdiv(mo, 256), 256, 0, stream>>>(out, blin, mo, OUTC);
  run_cvt(sx, x16, (long long)NS * HC, stream);
  run_wmma(x16, w16, out, NS, OUTC, stream);
}
